// TiLISTA_43971875176485
// MI455X (gfx1250) — compile-verified, matrix operand order checked by reading, not run on hardware
//
#include <hip/hip_runtime.h>
#include <hip/hip_bf16.h>
#include <cstdint>

#define M_DIM 512
#define N_DIM 2048
#define B_DIM 8192
#define T_STEPS 5

typedef __attribute__((ext_vector_type(16))) __bf16 v16bf;
typedef __attribute__((ext_vector_type(8)))  float  v8f;

union FragBF {
    v16bf v;
    uint4 q[2];
};

// ---- CDNA5 async global->LDS copy (ASYNCcnt path), 16 bytes per lane ----
__device__ __forceinline__ void async_copy_b128(unsigned ldsByteAddr, const void* gaddr) {
    asm volatile("global_load_async_to_lds_b128 %0, %1, off"
                 :: "v"(ldsByteAddr), "v"(gaddr)
                 : "memory");
}
__device__ __forceinline__ void wait_async0() {
    asm volatile("s_wait_asynccnt 0x0" ::: "memory");
}
// Flat shared-aperture pointer -> 32-bit LDS offset (ISA 10.2: the LDS offset is the low 32 address bits)
__device__ __forceinline__ unsigned lds_addr(const void* p) {
    return (unsigned)(uintptr_t)p;
}

__global__ void f32_to_bf16_kernel(const float* __restrict__ in,
                                   __hip_bfloat16* __restrict__ out, int n) {
    int i = blockIdx.x * blockDim.x + threadIdx.x;
    if (i < n) out[i] = __float2bfloat16(in[i]);
}

// out[b,c] = sum_k X[b,k] * Bop[c,k]   (X bf16 [Brows][K], Bop bf16 [Ncols][K], row-major)
// MODE 0: outF = mu[muIdx]*acc ; outBf = bf16(same)
// MODE 1: outBf = bf16(acc - yPtr[..])
// MODE 2: outF  = xoldF[..] - mu[muIdx]*acc  (in place safe: 1 read then 1 write per element)
template <int MODE>
__global__ __launch_bounds__(256)
void gemm_bf16_wmma(const __hip_bfloat16* __restrict__ X,
                    const __hip_bfloat16* __restrict__ Bop,
                    int K, int Ncols,
                    const float* __restrict__ muPtr, int muIdx,
                    const float* __restrict__ yPtr,
                    const float* __restrict__ xoldF,
                    float* __restrict__ outF,
                    __hip_bfloat16* __restrict__ outBf)
{
    // Block tile 128(rows) x 128(cols), K-step 32. Double-buffered tiles.
    // Row stride 40 shorts (80B): 16B-aligned, and 80*m mod 256 covers all 64 banks.
    __shared__ __align__(16) unsigned short ldsX[2][128 * 40];
    __shared__ __align__(16) unsigned short ldsW[2][128 * 40];

    const int tid  = threadIdx.x;
    const int lane = tid & 31;
    const int wave = tid >> 5;
    const int m16  = lane & 15;
    const int h    = lane >> 4;          // half-wave selector of the 16-bit WMMA layout
    const int waveRowBase = (wave >> 2) * 64;  // 2 wave-rows of 64
    const int waveColBase = (wave & 3) * 32;   // 4 wave-cols of 32

    const int cBase = blockIdx.x * 128;
    const int bBase = blockIdx.y * 128;

    // Staging: 256 threads cover 128 rows x 32 k; thread -> (row tid>>1, 16-short half tid&1)
    const int ldRow = tid >> 1;
    const int ldSeg = tid & 1;

    const unsigned short* Xg = (const unsigned short*)X;
    const unsigned short* Wg = (const unsigned short*)Bop;

    v8f acc[4][2];
    #pragma unroll
    for (int i = 0; i < 4; ++i)
        #pragma unroll
        for (int j = 0; j < 2; ++j)
            acc[i][j] = v8f{};

    auto stage = [&](int k0, int nb) {
        const unsigned short* gx = Xg + (size_t)(bBase + ldRow) * K + k0 + ldSeg * 16;
        const unsigned lx = lds_addr(&ldsX[nb][ldRow * 40 + ldSeg * 16]);
        async_copy_b128(lx,      gx);
        async_copy_b128(lx + 16, gx + 8);
        const unsigned short* gw = Wg + (size_t)(cBase + ldRow) * K + k0 + ldSeg * 16;
        const unsigned lw = lds_addr(&ldsW[nb][ldRow * 40 + ldSeg * 16]);
        async_copy_b128(lw,      gw);
        async_copy_b128(lw + 16, gw + 8);
    };

    stage(0, 0);
    wait_async0();
    __syncthreads();

    int cur = 0;
    for (int k0 = 0; k0 < K; k0 += 32) {
        const int nxt = cur ^ 1;
        if (k0 + 32 < K) {
            stage(k0 + 32, nxt);                 // async into other buffer, overlapped
            if (k0 + 64 < K) {                   // L2 pre-warm two tiles ahead
                __builtin_prefetch(Xg + (size_t)(bBase + ldRow) * K + (k0 + 64) + ldSeg * 16, 0, 3);
                __builtin_prefetch(Wg + (size_t)(cBase + ldRow) * K + (k0 + 64) + ldSeg * 16, 0, 3);
            }
        }

        // Fragments: A 16x32 per 16-row slice; B 32x16 per 16-col slice ([col][k] tile)
        FragBF a[4], b[2];
        #pragma unroll
        for (int i = 0; i < 4; ++i) {
            const unsigned short* p = &ldsX[cur][(waveRowBase + 16 * i + m16) * 40 + 8 * h];
            a[i].q[0] = *(const uint4*)(p);
            a[i].q[1] = *(const uint4*)(p + 16);
        }
        #pragma unroll
        for (int j = 0; j < 2; ++j) {
            const unsigned short* p = &ldsW[cur][(waveColBase + 16 * j + m16) * 40 + 8 * h];
            b[j].q[0] = *(const uint4*)(p);
            b[j].q[1] = *(const uint4*)(p + 16);
        }
        #pragma unroll
        for (int i = 0; i < 4; ++i)
            #pragma unroll
            for (int j = 0; j < 2; ++j)
                acc[i][j] = __builtin_amdgcn_wmma_f32_16x16x32_bf16(
                    false, a[i].v, false, b[j].v, (short)0, acc[i][j], false, false);

        wait_async0();
        __syncthreads();
        cur = nxt;
    }

    // C/D layout: VGPR e -> row e (lanes 0-15) / row e+8 (lanes 16-31), col = lane%16
    const float mu = (MODE != 1) ? muPtr[muIdx] : 0.0f;
    #pragma unroll
    for (int i = 0; i < 4; ++i) {
        #pragma unroll
        for (int j = 0; j < 2; ++j) {
            const int n  = cBase + waveColBase + 16 * j + m16;
            const int r0 = bBase + waveRowBase + 16 * i + 8 * h;
            #pragma unroll
            for (int e = 0; e < 8; ++e) {
                const size_t off = (size_t)(r0 + e) * Ncols + n;
                const float v = acc[i][j][e];
                if (MODE == 0) {
                    const float o = mu * v;
                    outF[off]  = o;
                    outBf[off] = __float2bfloat16(o);
                } else if (MODE == 1) {
                    const float o = v - yPtr[off];
                    outBf[off] = __float2bfloat16(o);
                } else {
                    const float o = xoldF[off] - mu * v;
                    outF[off] = o;
                }
            }
        }
    }
}

// One block per row of x_pre [B, N_DIM]: 4-pass MSB-first radix select of the
// kSel-th largest |x| (|float| bits order-match as uint), then
// x = (|x| >= thr) ? x : sign(x)*max(|x|-beta, 0). Writes fp32 + bf16 copies.
__global__ __launch_bounds__(256)
void shrink_topk_kernel(const float* __restrict__ xpre,
                        const float* __restrict__ betaPtr, int betaIdx,
                        int kSel,
                        float* __restrict__ xOutF,
                        __hip_bfloat16* __restrict__ xOutBf)
{
    __shared__ unsigned hist[256];
    __shared__ unsigned selPrefix;
    __shared__ unsigned selK;

    const int tid = threadIdx.x;
    const size_t rowOff = (size_t)blockIdx.x * N_DIM;

    float    v[8];
    unsigned key[8];
    #pragma unroll
    for (int i = 0; i < 8; ++i) {
        v[i]   = xpre[rowOff + tid + i * 256];
        key[i] = __float_as_uint(v[i]) & 0x7fffffffu;
    }

    unsigned prefix = 0;
    unsigned k = (unsigned)kSel;
    #pragma unroll
    for (int pass = 0; pass < 4; ++pass) {
        const int shift = 24 - 8 * pass;
        hist[tid] = 0;
        __syncthreads();
        const unsigned maskHi = (pass == 0) ? 0u : (0xffffffffu << (shift + 8));
        #pragma unroll
        for (int i = 0; i < 8; ++i) {
            if ((key[i] & maskHi) == (prefix & maskHi))
                atomicAdd(&hist[(key[i] >> shift) & 255], 1u);
        }
        __syncthreads();
        if (tid == 0) {
            unsigned cum = 0, bin = 0, kk = k;
            for (int b = 255; b >= 0; --b) {
                cum += hist[b];
                if (cum >= k) { bin = (unsigned)b; kk = k - (cum - hist[b]); break; }
            }
            selPrefix = prefix | (bin << shift);
            selK = kk;
        }
        __syncthreads();
        prefix = selPrefix;
        k = selK;
        __syncthreads();
    }

    const float thr = __uint_as_float(prefix);
    const float bta = betaPtr[betaIdx];
    #pragma unroll
    for (int i = 0; i < 8; ++i) {
        const float av   = fabsf(v[i]);
        const float soft = copysignf(fmaxf(av - bta, 0.0f), v[i]);
        const float o    = (av >= thr) ? v[i] : soft;
        xOutF[rowOff + tid + i * 256]  = o;
        xOutBf[rowOff + tid + i * 256] = __float2bfloat16(o);
    }
}

extern "C" void kernel_launch(void* const* d_in, const int* in_sizes, int n_in,
                              void* d_out, int out_size, void* d_ws, size_t ws_size,
                              hipStream_t stream)
{
    const float* y    = (const float*)d_in[0];  // [B, M]
    const float* A    = (const float*)d_in[1];  // [M, N]
    const float* W    = (const float*)d_in[2];  // [N, M] == A^T
    const float* beta = (const float*)d_in[3];  // [T+1]
    const float* mu   = (const float*)d_in[4];  // [T+1]
    float* out = (float*)d_out;                 // [B, N]

    char* ws = (char*)d_ws;
    size_t off = 0;
    auto alloc = [&](size_t bytes) -> char* {
        char* p = ws + off;
        off += (bytes + 255) & ~(size_t)255;
        return p;
    };
    __hip_bfloat16* Abf  = (__hip_bfloat16*)alloc((size_t)M_DIM * N_DIM * 2); // rows of A
    __hip_bfloat16* AbfT = (__hip_bfloat16*)alloc((size_t)N_DIM * M_DIM * 2); // rows of A^T
    __hip_bfloat16* ybf  = (__hip_bfloat16*)alloc((size_t)B_DIM * M_DIM * 2);
    __hip_bfloat16* xBf  = (__hip_bfloat16*)alloc((size_t)B_DIM * N_DIM * 2);
    __hip_bfloat16* rBf  = (__hip_bfloat16*)alloc((size_t)B_DIM * M_DIM * 2);
    float*          xF   = (float*)alloc((size_t)B_DIM * N_DIM * 4);

    f32_to_bf16_kernel<<<(M_DIM * N_DIM + 255) / 256, 256, 0, stream>>>(A, Abf, M_DIM * N_DIM);
    f32_to_bf16_kernel<<<(N_DIM * M_DIM + 255) / 256, 256, 0, stream>>>(W, AbfT, N_DIM * M_DIM);
    f32_to_bf16_kernel<<<(B_DIM * M_DIM + 255) / 256, 256, 0, stream>>>(y, ybf, B_DIM * M_DIM);

    dim3 blk(256);
    dim3 gN(N_DIM / 128, B_DIM / 128);  // output [B, N]
    dim3 gM(M_DIM / 128, B_DIM / 128);  // output [B, M]

    // x0 = mu[0] * (y @ A); t=0 shrink is identity (k == 0)
    gemm_bf16_wmma<0><<<gN, blk, 0, stream>>>(ybf, AbfT, M_DIM, N_DIM, mu, 0,
                                              nullptr, nullptr, xF, xBf);

    for (int t = 1; t <= T_STEPS; ++t) {
        // r = x @ A^T - y              [B, M]
        gemm_bf16_wmma<1><<<gM, blk, 0, stream>>>(xBf, Abf, N_DIM, M_DIM, nullptr, 0,
                                                  y, nullptr, nullptr, rBf);
        // x_pre = x - mu[t] * (r @ A)  (in place in xF)
        gemm_bf16_wmma<2><<<gN, blk, 0, stream>>>(rBf, AbfT, M_DIM, N_DIM, mu, t,
                                                  nullptr, xF, xF, nullptr);
        // row-wise top-k threshold + softshrink fallback
        double frac = 0.012 * t;
        if (frac > 0.12) frac = 0.12;
        const int kSel = (int)(frac * N_DIM);   // 24, 49, 73, 98, 122
        shrink_topk_kernel<<<B_DIM, 256, 0, stream>>>(xF, beta, t, kSel,
                                                      (t == T_STEPS) ? out : xF, xBf);
    }

    (void)in_sizes; (void)n_in; (void)out_size; (void)ws_size;
}
